// StructuralAttentionLayer_120259085229
// MI455X (gfx1250) — matrix operand order repaired and checked by reading, not compiled, not run on hardware
//
#include <hip/hip_runtime.h>

#define NN 6144
#define DD 128
#define NEG_INF_F (-1e12f)
#define MASK_WORDS_PER_ROW (NN / 32)

typedef __attribute__((ext_vector_type(16))) _Float16 v16h;
typedef __attribute__((ext_vector_type(8)))  float    v8f;

// ---------------- small prep kernels ----------------

__global__ void k_zero_mask(unsigned* __restrict__ mask, int nwords) {
    int i = blockIdx.x * 256 + threadIdx.x;
    if (i < nwords) mask[i] = 0u;
}

// u1[d] = sum_k W[d][k]*a[k];  u2[d] = sum_k W[d][k]*a[128+k]
__global__ void k_u(const float* __restrict__ W, const float* __restrict__ a,
                    float* __restrict__ u) {
    int d = threadIdx.x;                 // 0..127
    float acc1 = 0.f, acc2 = 0.f;
    for (int k = 0; k < DD; ++k) {
        float w = W[d * DD + k];
        acc1 = fmaf(w, a[k], acc1);
        acc2 = fmaf(w, a[DD + k], acc2);
    }
    u[d] = acc1;
    u[DD + d] = acc2;
}

// s1[n] = x[n,:].u1 ; s2[n] = x[n,:].u2   (one wave per row)
__global__ __launch_bounds__(256) void k_s(const float* __restrict__ x,
                                           const float* __restrict__ u,
                                           float* __restrict__ s1,
                                           float* __restrict__ s2) {
    int wave = threadIdx.x >> 5, lane = threadIdx.x & 31;
    int row = blockIdx.x * 8 + wave;
    float a1 = 0.f, a2 = 0.f;
    for (int t = lane; t < DD; t += 32) {
        float xv = x[(size_t)row * DD + t];
        a1 = fmaf(xv, u[t], a1);
        a2 = fmaf(xv, u[DD + t], a2);
    }
    for (int off = 16; off; off >>= 1) {
        a1 += __shfl_xor(a1, off, 32);
        a2 += __shfl_xor(a2, off, 32);
    }
    if (lane == 0) { s1[row] = a1; s2[row] = a2; }
}

// xT[d][n] = (f16) x[n][d]   (B-matrix feed for WMMA, column-contiguous)
__global__ void k_xT(const float* __restrict__ x, _Float16* __restrict__ xT) {
    int idx = blockIdx.x * 256 + threadIdx.x;     // 0 .. D*N-1
    if (idx >= DD * NN) return;
    int d = idx / NN, n = idx - d * NN;
    xT[idx] = (_Float16)x[(size_t)n * DD + d];
}

// mask bit (i,j) |= (edge_weight>0)  — exact for non-negative weights
__global__ void k_scatter(const int* __restrict__ ei, const float* __restrict__ w,
                          unsigned* __restrict__ mask, int E) {
    int e = blockIdx.x * 256 + threadIdx.x;
    if (e >= E) return;
    if (w[e] > 0.f) {
        int i = ei[e], j = ei[E + e];
        atomicOr(&mask[(size_t)i * MASK_WORDS_PER_ROW + (j >> 5)], 1u << (j & 31));
    }
}

// ---------------- fused masked dual-softmax + WMMA (attention @ x) ----------------
// block = 256 threads (8 waves) handles 16 rows; wave w owns D-columns [16w,16w+16)
// LDS P-tiles are double-buffered: one barrier per 32-column chunk.

__global__ __launch_bounds__(256) void k_attn(const _Float16* __restrict__ xT,
                                              const unsigned* __restrict__ mask,
                                              const float* __restrict__ s1g,
                                              const float* __restrict__ s2g,
                                              float* __restrict__ hp,
                                              float* __restrict__ hpn) {
    const int tid  = threadIdx.x;
    const int r    = tid >> 4;       // row in tile 0..15
    const int g    = tid & 15;       // column-group 0..15
    const int lane = tid & 31;
    const int wave = tid >> 5;       // 0..7
    const int row0 = blockIdx.x * 16;

    __shared__ float sm_s1[16];
    __shared__ float sm_m[2][16], sm_iZ[2][16];
    __shared__ __align__(32) _Float16 smA[2][2][16 * 32]; // [buf][sign][A-layout]

    if (tid < 16) sm_s1[tid] = s1g[row0 + tid];
    __syncthreads();

    const float s1r = sm_s1[r];
    const unsigned* mrow = mask + (size_t)(row0 + r) * MASK_WORDS_PER_ROW;

    // ---- pass 1: online (max, sumexp) for +e and -e, literal NEG_INF fill ----
    float mp = -3.0e38f, Zp = 0.f, mn = -3.0e38f, Zn = 0.f;
    for (int j = g; j < NN; j += 16) {
        float e = s1r + s2g[j];
        e = e > 0.f ? e : 0.2f * e;                       // leaky_relu(0.2)
        bool mm = (mrow[j >> 5] >> (j & 31)) & 1u;
        float ep = mm ? e  : NEG_INF_F;
        float en = mm ? -e : NEG_INF_F;
        if (ep > mp) { Zp = Zp * __expf(mp - ep) + 1.f; mp = ep; }
        else         { Zp += __expf(ep - mp); }
        if (en > mn) { Zn = Zn * __expf(mn - en) + 1.f; mn = en; }
        else         { Zn += __expf(en - mn); }
    }
    // combine 16 partials per row (xor within 16-lane groups stays in-wave)
    for (int off = 1; off < 16; off <<= 1) {
        float mo = __shfl_xor(mp, off, 32), Zo = __shfl_xor(Zp, off, 32);
        float mx = fmaxf(mp, mo);
        Zp = Zp * __expf(mp - mx) + Zo * __expf(mo - mx); mp = mx;
        mo = __shfl_xor(mn, off, 32); Zo = __shfl_xor(Zn, off, 32);
        mx = fmaxf(mn, mo);
        Zn = Zn * __expf(mn - mx) + Zo * __expf(mo - mx); mn = mx;
    }
    if (g == 0) {
        sm_m[0][r] = mp; sm_iZ[0][r] = 1.f / Zp;
        sm_m[1][r] = mn; sm_iZ[1][r] = 1.f / Zn;
    }
    __syncthreads();

    const float mpr = sm_m[0][r], iZp = sm_iZ[0][r];
    const float mnr = sm_m[1][r], iZn = sm_iZ[1][r];

    // ---- pass 2: build P tiles (f16) and accumulate via WMMA ----
    v8f cpos = {}; v8f cneg = {};
    const int colw = wave * 16 + (lane & 15);
    const _Float16* xTcol = xT + (size_t)colw * NN + 16 * (lane >> 4);

    // A-layout target slot for this thread: k = 2g (pair k,k+1 packed as one dword), lane half = k/16, slot = k%16
    const int k    = 2 * g;
    const int lw   = r + 16 * (k >> 4);
    const int slot = k & 15;

    for (int c = 0; c < NN / 32; ++c) {
        const int j0  = c * 32;
        const int buf = c & 1;
        const int j   = j0 + k;
        float e0 = s1r + s2g[j];     e0 = e0 > 0.f ? e0 : 0.2f * e0;
        float e1 = s1r + s2g[j + 1]; e1 = e1 > 0.f ? e1 : 0.2f * e1;
        unsigned mw = mrow[j0 >> 5];
        bool m0 = (mw >> k) & 1u, m1 = (mw >> (k + 1)) & 1u;
        float p0 =  __expf((m0 ?  e0 : NEG_INF_F) - mpr) * iZp;
        float p1 =  __expf((m1 ?  e1 : NEG_INF_F) - mpr) * iZp;
        float q0 = -__expf((m0 ? -e0 : NEG_INF_F) - mnr) * iZn;   // fold sign of neg-attention
        float q1 = -__expf((m1 ? -e1 : NEG_INF_F) - mnr) * iZn;

        union { _Float16 h[2]; unsigned u; } pk;
        pk.h[0] = (_Float16)p0; pk.h[1] = (_Float16)p1;
        *(unsigned*)(&smA[buf][0][lw * 16 + slot]) = pk.u;
        pk.h[0] = (_Float16)q0; pk.h[1] = (_Float16)q1;
        *(unsigned*)(&smA[buf][1][lw * 16 + slot]) = pk.u;
        __syncthreads();   // also protects WAR on this buffer (last read at c-2)

        v16h a_p = *(const v16h*)(&smA[buf][0][lane * 16]);
        v16h a_n = *(const v16h*)(&smA[buf][1][lane * 16]);
        v16h b   = *(const v16h*)(xTcol + j0);  // B: lanes0-15 K=0-15, lanes16-31 K=16-31

        cpos = __builtin_amdgcn_wmma_f32_16x16x32_f16(false, a_p, false, b,
                                                      (short)0, cpos, false, false);
        cneg = __builtin_amdgcn_wmma_f32_16x16x32_f16(false, a_n, false, b,
                                                      (short)0, cneg, false, false);
    }

    // C/D layout: VGPR rr -> M=rr (lanes0-15) / M=rr+8 (lanes16-31), N=lane&15
#pragma unroll
    for (int rr = 0; rr < 8; ++rr) {
        int row = row0 + rr + 8 * (lane >> 4);
        hp [(size_t)row * DD + colw] = cpos[rr];
        hpn[(size_t)row * DD + colw] = cneg[rr];
    }
}

// ---------------- out = elu(concat(hp,hpn) @ wtrans) ----------------
// block = 128 threads handles 8 rows
__global__ __launch_bounds__(128) void k_final(const float* __restrict__ hp,
                                               const float* __restrict__ hpn,
                                               const float* __restrict__ wt,
                                               float* __restrict__ out) {
    __shared__ float sm[8][256];
    const int r0  = blockIdx.x * 8;
    const int col = threadIdx.x;                 // 0..127
    for (int t = threadIdx.x; t < 8 * 256; t += 128) {
        int rr = t >> 8, c = t & 255;
        sm[rr][c] = (c < DD) ? hp[(size_t)(r0 + rr) * DD + c]
                             : hpn[(size_t)(r0 + rr) * DD + (c - DD)];
    }
    __syncthreads();
    float acc[8] = {};
    for (int c = 0; c < 2 * DD; ++c) {
        float wv = wt[(size_t)c * DD + col];
#pragma unroll
        for (int rr = 0; rr < 8; ++rr) acc[rr] = fmaf(sm[rr][c], wv, acc[rr]);
    }
#pragma unroll
    for (int rr = 0; rr < 8; ++rr) {
        float v = acc[rr];
        out[(size_t)(r0 + rr) * DD + col] = v > 0.f ? v : expm1f(v);
    }
}

// ---------------- host launcher ----------------

extern "C" void kernel_launch(void* const* d_in, const int* in_sizes, int n_in,
                              void* d_out, int out_size, void* d_ws, size_t ws_size,
                              hipStream_t stream) {
    const float* x   = (const float*)d_in[0];   // [N,128]
    const float* ew  = (const float*)d_in[1];   // [E]
    const float* W   = (const float*)d_in[2];   // [128,128]
    const float* a   = (const float*)d_in[3];   // [256]
    const float* wt  = (const float*)d_in[4];   // [256,128]
    const int*   ei  = (const int*)d_in[5];     // [2,E]
    float* out = (float*)d_out;
    const int E = in_sizes[1];

    uint8_t* ws = (uint8_t*)d_ws;
    const size_t MASK_BYTES = (size_t)NN * MASK_WORDS_PER_ROW * 4;   // 4,718,592
    unsigned*  mask = (unsigned*)ws;
    _Float16*  xT   = (_Float16*)(ws + MASK_BYTES);                  // 1,572,864
    float*     u    = (float*)(ws + MASK_BYTES + 1572864);           // 1,024
    float*     s1   = (float*)(ws + MASK_BYTES + 1572864 + 1024);
    float*     s2   = (float*)(ws + MASK_BYTES + 1572864 + 1024 + 24576);
    float*     hp   = (float*)(ws + MASK_BYTES + 1572864 + 1024 + 2 * 24576);
    float*     hpn  = (float*)(ws + MASK_BYTES + 1572864 + 1024 + 2 * 24576 + 3145728);

    const int nwords = NN * MASK_WORDS_PER_ROW;                      // 1,179,648
    k_zero_mask<<<(nwords + 255) / 256, 256, 0, stream>>>(mask, nwords);
    k_u<<<1, 128, 0, stream>>>(W, a, u);
    k_xT<<<(DD * NN + 255) / 256, 256, 0, stream>>>(x, xT);
    k_s<<<NN / 8, 256, 0, stream>>>(x, u, s1, s2);
    k_scatter<<<(E + 255) / 256, 256, 0, stream>>>(ei, ew, mask, E);
    k_attn<<<NN / 16, 256, 0, stream>>>(xT, mask, s1, s2, hp, hpn);
    k_final<<<NN / 8, 128, 0, stream>>>(hp, hpn, wt, out);
}
